// GNNMOT_72043781423310
// MI455X (gfx1250) — compile-verified, matrix operand order checked by reading, not run on hardware
//
#include <hip/hip_runtime.h>
#include <hip/hip_bf16.h>
#include <math.h>

typedef __attribute__((ext_vector_type(16))) _Float16 v16h;
typedef __attribute__((ext_vector_type(8)))  float    v8f;
typedef __attribute__((ext_vector_type(4)))  int      v4i;

#define AS1 __attribute__((address_space(1)))
#define AS3 __attribute__((address_space(3)))

#if __has_builtin(__builtin_amdgcn_global_load_async_to_lds_b128) && \
    __has_builtin(__builtin_amdgcn_s_wait_asynccnt)
#define USE_ASYNC_LDS 1
#else
#define USE_ASYNC_LDS 0
#endif

// ---------------------------------------------------------------------------
// WMMA fragment loaders (CDNA5 16-bit A/B layout, wave32):
//   lanes 0-15 : row/col = lane,    K in {0..7} U {16..23}
//   lanes 16-31: row/col = lane-16, K in {8..15} U {24..31}
// Source is an [outer][K] array (A: outer=M rows; B: outer=N cols of W[N][K]).
// ---------------------------------------------------------------------------
__device__ inline v16h load_frag_h16(const _Float16* t, int ld, int lane) {
  int row = lane & 15;
  int kb  = (lane < 16) ? 0 : 8;
  const _Float16* p = t + row * ld + kb;
  v16h r;
#pragma unroll
  for (int i = 0; i < 8; ++i) { r[i] = p[i]; r[i + 8] = p[16 + i]; }
  return r;
}

__device__ inline v16h load_frag_f32(const float* t, int ld, int lane) {
  int row = lane & 15;
  int kb  = (lane < 16) ? 0 : 8;
  const float* p = t + row * ld + kb;
  v16h r;
#pragma unroll
  for (int i = 0; i < 8; ++i) {
    r[i]     = (_Float16)p[i];
    r[i + 8] = (_Float16)p[16 + i];
  }
  return r;
}

__device__ inline v8f wmma_f16(v16h a, v16h b, v8f c) {
  return __builtin_amdgcn_wmma_f32_16x16x32_f16(false, a, false, b, (short)0, c,
                                                false, false);
}

__device__ inline float sigmoidf_(float x) { return 1.f / (1.f + expf(-x)); }

// ---------------------------------------------------------------------------
// f32 -> f16 conversion (weights, once per launch)
// ---------------------------------------------------------------------------
__global__ void __launch_bounds__(256)
cvt_f16_kernel(const float* __restrict__ src, _Float16* __restrict__ dst, int n) {
  int i = blockIdx.x * 256 + threadIdx.x;
  if (i < n) dst[i] = (_Float16)src[i];
}

// ---------------------------------------------------------------------------
// PointNet: per-box shared MLP 3->64->128->128 + max over 256 points.
// One box per block (1024 blocks: 0..511 det, 512..1023 trk), 256 thr = 8 waves.
// Writes the 128-dim appearance feature (f16) into h[box*256 + 0..127].
// ---------------------------------------------------------------------------
__global__ void __launch_bounds__(256)
pointnet_kernel(const float* __restrict__ det_pc, const float* __restrict__ trk_pc,
                const float* __restrict__ w1, const float* __restrict__ b1,
                const float* __restrict__ w2, const float* __restrict__ b2,
                const float* __restrict__ w3, const float* __restrict__ b3,
                _Float16* __restrict__ hout) {
  __shared__ _Float16 H1[128][72];    // 128x64 layer-1 activations (padded)
  __shared__ _Float16 H2[128][136];   // 128x128 layer-2 activations (padded)

  const int box  = blockIdx.x;
  const float* pc = (box < 512) ? det_pc + (size_t)box * 256 * 3
                                : trk_pc + (size_t)(box - 512) * 256 * 3;
  const int lane = threadIdx.x & 31;
  const int wave = threadIdx.x >> 5;   // 0..7

  float vmax[8];
#pragma unroll
  for (int r = 0; r < 8; ++r) vmax[r] = -3.4e38f;

  for (int chunk = 0; chunk < 2; ++chunk) {
    const float* pcc = pc + chunk * 128 * 3;

    // layer 1 (K=3) in VALU -> LDS f16
    for (int i = threadIdx.x; i < 128 * 64; i += 256) {
      int p = i >> 6, o = i & 63;
      float s = b1[o] + w1[o * 3 + 0] * pcc[p * 3 + 0] +
                w1[o * 3 + 1] * pcc[p * 3 + 1] + w1[o * 3 + 2] * pcc[p * 3 + 2];
      H1[p][o] = (_Float16)fmaxf(s, 0.f);
    }
    __syncthreads();

    // layer 2: [128x64] @ w2[128x64]^T -> [128x128], WMMA, K=64
    {
      const int rt = wave;  // row tile 0..7
      for (int ct = 0; ct < 8; ++ct) {
        v8f acc = {};
#pragma unroll
        for (int k0 = 0; k0 < 64; k0 += 32) {
          v16h a = load_frag_h16(&H1[rt * 16][k0], 72, lane);
          v16h b = load_frag_f32(w2 + (ct * 16) * 64 + k0, 64, lane);
          acc = wmma_f16(a, b, acc);
        }
        const int cn = lane & 15, mh = (lane >> 4) * 8;
        const int gcol = ct * 16 + cn;
        const float bb = b2[gcol];
#pragma unroll
        for (int r = 0; r < 8; ++r)
          H2[rt * 16 + mh + r][gcol] = (_Float16)fmaxf(acc[r] + bb, 0.f);
      }
    }
    __syncthreads();

    // layer 3: [128x128] @ w3[128x128]^T, WMMA, K=128; fold relu+max into epilogue
    {
      const int ct = wave;  // col tile 0..7
      for (int rt = 0; rt < 8; ++rt) {
        v8f acc = {};
#pragma unroll
        for (int k0 = 0; k0 < 128; k0 += 32) {
          v16h a = load_frag_h16(&H2[rt * 16][k0], 136, lane);
          v16h b = load_frag_f32(w3 + (ct * 16) * 128 + k0, 128, lane);
          acc = wmma_f16(a, b, acc);
        }
        const int cn = lane & 15;
        const float bb = b3[ct * 16 + cn];
#pragma unroll
        for (int r = 0; r < 8; ++r)
          vmax[r] = fmaxf(vmax[r], fmaxf(acc[r] + bb, 0.f));
      }
    }
    __syncthreads();
  }

  float m = vmax[0];
#pragma unroll
  for (int r = 1; r < 8; ++r) m = fmaxf(m, vmax[r]);
  m = fmaxf(m, __shfl_xor(m, 16, 32));  // merge M-halves (same column)
  if (lane < 16) {
    int col = wave * 16 + lane;  // 0..127
    hout[(size_t)box * 256 + col] = (_Float16)m;
  }
}

// ---------------------------------------------------------------------------
// Det motion MLP: 9 -> relu 64 -> 128 (no final relu). One det per block.
// ---------------------------------------------------------------------------
__global__ void __launch_bounds__(128)
detmot_kernel(const float* __restrict__ boxes, const float* __restrict__ w1,
              const float* __restrict__ b1, const float* __restrict__ w2,
              const float* __restrict__ b2, _Float16* __restrict__ h0) {
  __shared__ float h1[64];
  const int s = blockIdx.x, tid = threadIdx.x;
  const float* x = boxes + s * 9;
  if (tid < 64) {
    float a = b1[tid];
#pragma unroll
    for (int k = 0; k < 9; ++k) a += w1[tid * 9 + k] * x[k];
    h1[tid] = fmaxf(a, 0.f);
  }
  __syncthreads();
  float a = b2[tid];
#pragma unroll
  for (int k = 0; k < 64; ++k) a += w2[tid * 64 + k] * h1[k];
  h0[(size_t)s * 256 + 128 + tid] = (_Float16)a;
}

// ---------------------------------------------------------------------------
// 2-layer LSTM (hidden 128, T=10, gate order i,f,g,o). One sequence per
// 128-thread block; thread t owns hidden unit t (4 gate rows).
// ---------------------------------------------------------------------------
__global__ void __launch_bounds__(128)
lstm_kernel(const float* __restrict__ xseq, const float* __restrict__ wih0,
            const float* __restrict__ whh0, const float* __restrict__ b0,
            const float* __restrict__ wih1, const float* __restrict__ whh1,
            const float* __restrict__ b1, _Float16* __restrict__ h0out) {
  __shared__ float hbuf[128], cbuf[128], xs0[10][128];
  const int seq = blockIdx.x, tid = threadIdx.x;

  // layer 0 (input dim 9)
  hbuf[tid] = 0.f; cbuf[tid] = 0.f;
  __syncthreads();
  for (int t = 0; t < 10; ++t) {
    const float* xt = xseq + ((size_t)seq * 10 + t) * 9;
    float g[4];
#pragma unroll
    for (int q = 0; q < 4; ++q) {
      const int row = q * 128 + tid;
      float s = b0[row];
#pragma unroll
      for (int k = 0; k < 9; ++k) s += wih0[row * 9 + k] * xt[k];
      for (int k = 0; k < 128; ++k) s += whh0[row * 128 + k] * hbuf[k];
      g[q] = s;
    }
    __syncthreads();
    const float cn = sigmoidf_(g[1]) * cbuf[tid] + sigmoidf_(g[0]) * tanhf(g[2]);
    const float hn = sigmoidf_(g[3]) * tanhf(cn);
    cbuf[tid] = cn; hbuf[tid] = hn; xs0[t][tid] = hn;
    __syncthreads();
  }

  // layer 1 (input dim 128)
  hbuf[tid] = 0.f; cbuf[tid] = 0.f;
  __syncthreads();
  for (int t = 0; t < 10; ++t) {
    float g[4];
#pragma unroll
    for (int q = 0; q < 4; ++q) {
      const int row = q * 128 + tid;
      float s = b1[row];
      for (int k = 0; k < 128; ++k) s += wih1[row * 128 + k] * xs0[t][k];
      for (int k = 0; k < 128; ++k) s += whh1[row * 128 + k] * hbuf[k];
      g[q] = s;
    }
    __syncthreads();
    const float cn = sigmoidf_(g[1]) * cbuf[tid] + sigmoidf_(g[0]) * tanhf(g[2]);
    const float hn = sigmoidf_(g[3]) * tanhf(cn);
    cbuf[tid] = cn; hbuf[tid] = hn;
    __syncthreads();
    if (t == 9) h0out[(size_t)(512 + seq) * 256 + 128 + tid] = (_Float16)hn;
  }
}

// ---------------------------------------------------------------------------
// WMMA GEMM (f16 in, f32 out): Out[MxN] = X[MxK] @ W[NxK]^T (+bias, opt relu).
// Block = 4 waves computing a 64x64 tile; each wave owns a 2x2 grid of 16x16
// accumulators (4 WMMAs / K-step, A/B fragments reused twice each).
// Tiles staged to LDS via CDNA5 async global->LDS (ASYNCcnt) when available.
// M,N multiples of 64; K multiple of 32.
// ---------------------------------------------------------------------------
__global__ void __launch_bounds__(128)
gemm_h16_kernel(const _Float16* __restrict__ X, const _Float16* __restrict__ Wt,
                const float* __restrict__ bias, float* __restrict__ Out,
                int M, int N, int K, int relu) {
  __shared__ _Float16 Xs[64][40];   // 40-halves stride keeps rows 16B-aligned
  __shared__ _Float16 Ws[64][40];
  const int lane = threadIdx.x & 31;
  const int wave = threadIdx.x >> 5;          // 0..3
  const int m0 = blockIdx.x * 64;
  const int n0 = blockIdx.y * 64;
  const int wm = (wave & 1) * 32;
  const int wn = (wave >> 1) * 32;

  v8f acc00 = {}, acc01 = {}, acc10 = {}, acc11 = {};

  for (int k0 = 0; k0 < K; k0 += 32) {
#if USE_ASYNC_LDS
    for (int i = threadIdx.x; i < 64 * 4; i += 128) {
      const int r = i >> 2, c = (i & 3) * 8;   // 8 halves = 16 bytes
      __builtin_amdgcn_global_load_async_to_lds_b128(
          (AS1 v4i*)(X + (size_t)(m0 + r) * K + k0 + c),
          (AS3 v4i*)&Xs[r][c], 0, 0);
      __builtin_amdgcn_global_load_async_to_lds_b128(
          (AS1 v4i*)(Wt + (size_t)(n0 + r) * K + k0 + c),
          (AS3 v4i*)&Ws[r][c], 0, 0);
    }
    __builtin_amdgcn_s_wait_asynccnt(0);
#else
    for (int i = threadIdx.x; i < 64 * 4; i += 128) {
      const int r = i >> 2, c = (i & 3) * 8;
      *(float4*)&Xs[r][c] = *(const float4*)(X + (size_t)(m0 + r) * K + k0 + c);
      *(float4*)&Ws[r][c] = *(const float4*)(Wt + (size_t)(n0 + r) * K + k0 + c);
    }
#endif
    __syncthreads();
    v16h a0 = load_frag_h16(&Xs[wm][0],      40, lane);
    v16h a1 = load_frag_h16(&Xs[wm + 16][0], 40, lane);
    v16h b0 = load_frag_h16(&Ws[wn][0],      40, lane);
    v16h b1 = load_frag_h16(&Ws[wn + 16][0], 40, lane);
    acc00 = wmma_f16(a0, b0, acc00);
    acc01 = wmma_f16(a0, b1, acc01);
    acc10 = wmma_f16(a1, b0, acc10);
    acc11 = wmma_f16(a1, b1, acc11);
    __syncthreads();
  }

  const int cn = lane & 15, mh = (lane >> 4) * 8;
  auto store_tile = [&](v8f acc, int rb, int cb) {
    const int gcol = n0 + cb + cn;
    const float bv = bias ? bias[gcol] : 0.f;
#pragma unroll
    for (int r = 0; r < 8; ++r) {
      float v = acc[r] + bv;
      if (relu) v = fmaxf(v, 0.f);
      Out[(size_t)(m0 + rb + mh + r) * N + gcol] = v;
    }
  };
  store_tile(acc00, wm,      wn);
  store_tile(acc01, wm,      wn + 16);
  store_tile(acc10, wm + 16, wn);
  store_tile(acc11, wm + 16, wn + 16);
}

// ---------------------------------------------------------------------------
// EdgeConv helpers: masked column max of A over source dets, then combine.
// ---------------------------------------------------------------------------
__global__ void __launch_bounds__(256)
col_max_kernel(const float* __restrict__ A, const float* __restrict__ aff,
               float* __restrict__ maxA) {
  const int m = blockIdx.x, o = threadIdx.x;  // 512 blocks x 256 threads
  float mv = -3.4e38f;
  for (int n = 0; n < 512; ++n) {
    const float a = aff[(size_t)n * 512 + m];          // broadcast per n
    const float v = A[(size_t)n * 256 + o];            // coalesced
    if (a > 0.f) mv = fmaxf(mv, v);
  }
  maxA[(size_t)m * 256 + o] = mv;
}

__global__ void __launch_bounds__(256)
combine_kernel(const float* __restrict__ phi, const float* __restrict__ maxA,
               const float* __restrict__ Bm, const float* __restrict__ tb,
               _Float16* __restrict__ hout) {
  const int i = blockIdx.x;   // node 0..1023
  const int o = threadIdx.x;  // channel 0..255
  const float p = phi[(size_t)i * 256 + o];
  const float t = tb[o];
  float v;
  if (i < 512) {
    v = p + t;                                         // dets: self loop only
  } else {
    const int m = i - 512;
    const float t1 = maxA[(size_t)m * 256 + o] - Bm[(size_t)m * 256 + o] + t;
    v = p + fmaxf(t, t1);                              // max(self-loop, edges)
  }
  hout[(size_t)i * 256 + o] = (_Float16)fmaxf(v, 0.f); // outer relu, store f16
}

// ---------------------------------------------------------------------------
// Edge regression head: e = relu(C[m]-D[n]) (b1 folded into C), sigmoid dot.
// ---------------------------------------------------------------------------
__global__ void __launch_bounds__(256)
edge_final_kernel(const float* __restrict__ Cm, const float* __restrict__ Dm,
                  const float* __restrict__ w2, const float* __restrict__ b2,
                  const float* __restrict__ aff, float* __restrict__ out) {
  __shared__ float Ds[64], W2s[64];
  const int n = blockIdx.x, tid = threadIdx.x;
  if (tid < 64) { Ds[tid] = Dm[(size_t)n * 64 + tid]; W2s[tid] = w2[tid]; }
  __syncthreads();
  for (int m = tid; m < 512; m += 256) {
    const float* cr = Cm + (size_t)m * 64;
    float s = b2[0];
#pragma unroll
    for (int k = 0; k < 64; ++k) s += fmaxf(cr[k] - Ds[k], 0.f) * W2s[k];
    const float r = sigmoidf_(s);
    const float v = aff[(size_t)n * 512 + m] * r;
    out[(size_t)n * 512 + m] = (v == 0.f) ? 99.f : v;
  }
}

// ---------------------------------------------------------------------------
extern "C" void kernel_launch(void* const* d_in, const int* in_sizes, int n_in,
                              void* d_out, int out_size, void* d_ws, size_t ws_size,
                              hipStream_t stream) {
  const float* det_pc   = (const float*)d_in[0];
  const float* det_box  = (const float*)d_in[1];
  const float* trk_pc   = (const float*)d_in[2];
  const float* trk_box  = (const float*)d_in[3];
  const float* aff      = (const float*)d_in[4];
  // d_in[5] = gt_affinity_matrix (unused in inference)
  const float* pn_w1 = (const float*)d_in[6];  const float* pn_b1 = (const float*)d_in[7];
  const float* pn_w2 = (const float*)d_in[8];  const float* pn_b2 = (const float*)d_in[9];
  const float* pn_w3 = (const float*)d_in[10]; const float* pn_b3 = (const float*)d_in[11];
  const float* mlp_w1 = (const float*)d_in[12]; const float* mlp_b1 = (const float*)d_in[13];
  const float* mlp_w2 = (const float*)d_in[14]; const float* mlp_b2 = (const float*)d_in[15];
  const float* l0_wih = (const float*)d_in[16]; const float* l0_whh = (const float*)d_in[17];
  const float* l0_b   = (const float*)d_in[18];
  const float* l1_wih = (const float*)d_in[19]; const float* l1_whh = (const float*)d_in[20];
  const float* l1_b   = (const float*)d_in[21];
  const float* ec_tw = (const float*)d_in[22]; const float* ec_tb = (const float*)d_in[23];
  const float* ec_pw = (const float*)d_in[24]; const float* ec_pb = (const float*)d_in[25];
  const float* er_w1 = (const float*)d_in[26]; const float* er_b1 = (const float*)d_in[27];
  const float* er_w2 = (const float*)d_in[28]; const float* er_b2 = (const float*)d_in[29];
  float* out = (float*)d_out;

  // workspace layout (bytes)
  char* base = (char*)d_ws;
  _Float16* h0   = (_Float16*)(base);                    // [1024x256] f16 512KB
  _Float16* h1   = (_Float16*)(base + (512 << 10));      // [1024x256] f16 512KB
  float*    phi  = (float*)   (base + (1024 << 10));     // [1024x256] f32 1MB
  float*    Amat = (float*)   (base + (2048 << 10));     // [512x256]  f32 512KB
  float*    Bmat = (float*)   (base + (2560 << 10));     // [512x256]  f32 512KB
  float*    maxA = (float*)   (base + (3072 << 10));     // [512x256]  f32 512KB
  float*    Cm   = (float*)   (base + (3584 << 10));     // [512x64]   f32 128KB
  float*    Dm   = (float*)   (base + (3712 << 10));     // [512x64]   f32 128KB
  _Float16* twH  = (_Float16*)(base + (3840 << 10));     // 4x256x256 f16 512KB
  _Float16* pwH  = (_Float16*)(base + (4352 << 10));     // 4x256x256 f16 512KB
  _Float16* ew1H = (_Float16*)(base + (4864 << 10));     // 64x256    f16 32KB

  // 0) one-time f16 weight conversion for the WMMA GEMM stages
  cvt_f16_kernel<<<(4 * 256 * 256 + 255) / 256, 256, 0, stream>>>(ec_tw, twH, 4 * 256 * 256);
  cvt_f16_kernel<<<(4 * 256 * 256 + 255) / 256, 256, 0, stream>>>(ec_pw, pwH, 4 * 256 * 256);
  cvt_f16_kernel<<<(64 * 256 + 255) / 256, 256, 0, stream>>>(er_w1, ew1H, 64 * 256);

  // 1) appearance features (WMMA pointnet) -> h0[:, 0:128]
  pointnet_kernel<<<1024, 256, 0, stream>>>(det_pc, trk_pc, pn_w1, pn_b1, pn_w2,
                                            pn_b2, pn_w3, pn_b3, h0);
  // 2) det motion MLP -> h0[0:512, 128:256]
  detmot_kernel<<<512, 128, 0, stream>>>(det_box, mlp_w1, mlp_b1, mlp_w2, mlp_b2, h0);
  // 3) track motion LSTM -> h0[512:1024, 128:256]
  lstm_kernel<<<512, 128, 0, stream>>>(trk_box, l0_wih, l0_whh, l0_b,
                                       l1_wih, l1_whh, l1_b, h0);

  // 4) four EdgeConv layers (factored): phi/A/B GEMMs + masked col-max + combine
  _Float16* hin = h0;
  _Float16* hnx = h1;
  for (int k = 0; k < 4; ++k) {
    const _Float16* tw = twH + (size_t)k * 256 * 256;
    const _Float16* pw = pwH + (size_t)k * 256 * 256;
    const float* tb = ec_tb + (size_t)k * 256;
    const float* pb = ec_pb + (size_t)k * 256;
    gemm_h16_kernel<<<dim3(16, 4), 128, 0, stream>>>(hin, pw, pb, phi,
                                                     1024, 256, 256, 0);
    gemm_h16_kernel<<<dim3(8, 4), 128, 0, stream>>>(hin, tw, nullptr, Amat,
                                                    512, 256, 256, 0);
    gemm_h16_kernel<<<dim3(8, 4), 128, 0, stream>>>(hin + (size_t)512 * 256, tw,
                                                    nullptr, Bmat, 512, 256, 256, 0);
    col_max_kernel<<<512, 256, 0, stream>>>(Amat, aff, maxA);
    combine_kernel<<<1024, 256, 0, stream>>>(phi, maxA, Bmat, tb, hnx);
    _Float16* t = hin; hin = hnx; hnx = t;
  }
  // after 4 swaps the final features are back in `hin` (== h0)

  // 5) edge regression head (factored): C = h_trk@w1^T + b1, D = h_det@w1^T
  gemm_h16_kernel<<<dim3(8, 1), 128, 0, stream>>>(hin + (size_t)512 * 256, ew1H,
                                                  er_b1, Cm, 512, 64, 256, 0);
  gemm_h16_kernel<<<dim3(8, 1), 128, 0, stream>>>(hin, ew1H, nullptr, Dm,
                                                  512, 64, 256, 0);
  edge_final_kernel<<<512, 256, 0, stream>>>(Cm, Dm, er_w2, er_b2, aff, out);
}
